// NormalAttention_60559038874129
// MI455X (gfx1250) — matrix operand order repaired and checked by reading, not compiled, not run on hardware
//
#include <hip/hip_runtime.h>
#include <hip/hip_bf16.h>
#include <stdint.h>

// ---------------------------------------------------------------------------
// Problem constants (reference: B=4, H=W=40, C=512, heads=8, dh=64)
// ---------------------------------------------------------------------------
#define BATCH   4
#define NPOS    1600          // H*W
#define C_IN    512
#define HEADS   8
#define DH      64
#define QK_O    1024          // q (512) + k (512) output features; v unused!

#define AS1 __attribute__((address_space(1)))
#define AS3 __attribute__((address_space(3)))

#if defined(__HIP_DEVICE_COMPILE__) && __has_builtin(__builtin_amdgcn_global_load_async_to_lds_b128)
#define USE_ASYNC_LDS 1
#else
#define USE_ASYNC_LDS 0
#endif

typedef __attribute__((ext_vector_type(16))) __bf16 v16bf;
typedef __attribute__((ext_vector_type(8)))  float  v8f;
typedef __attribute__((vector_size(16)))     int    v4i;   // gcc-vector int4

union Frag16 {
    v16bf          v;
    uint32_t       w[8];
    unsigned short u[16];
};

__device__ __forceinline__ unsigned short f32_to_bf16(float f) {
    union { float f; uint32_t u; } x; x.f = f;
    uint32_t r = x.u + 0x7FFFu + ((x.u >> 16) & 1u);   // round-to-nearest-even
    return (unsigned short)(r >> 16);
}

__device__ __forceinline__ v8f wmma_bf16(const v16bf a, const v16bf b, v8f c) {
    return __builtin_amdgcn_wmma_f32_16x16x32_bf16(
        /*neg_a=*/false, a, /*neg_b=*/false, b,
        /*c_mod=*/(short)0, c, /*reuse_a=*/false, /*reuse_b=*/false);
}

// ---------------------------------------------------------------------------
// Kernel 0: fp32 -> bf16 conversion (grid-stride)
// ---------------------------------------------------------------------------
__global__ void cvt_bf16_kernel(const float* __restrict__ src,
                                unsigned short* __restrict__ dst, int n) {
    int i = blockIdx.x * blockDim.x + threadIdx.x;
    int stride = gridDim.x * blockDim.x;
    for (; i < n; i += stride) dst[i] = f32_to_bf16(src[i]);
}

// ---------------------------------------------------------------------------
// Kernel 1: qk = x @ w[0:1024]^T + b, then RoPE; q gets dh^-0.5 folded in.
// One wave computes a 16(m) x 64(o) tile. 6400 wave tasks, 8 waves/block.
// Output layout: q_bf/k_bf as (b*heads, n, dh) bf16.
// ---------------------------------------------------------------------------
__global__ __launch_bounds__(256) void qk_proj_rope_kernel(
    const unsigned short* __restrict__ xb,     // (B*N, C) bf16
    const unsigned short* __restrict__ wb,     // (1024, C) bf16
    const float* __restrict__ bias,            // (1536,)
    const float* __restrict__ sinp,            // (N, DH)
    const float* __restrict__ cosp,            // (N, DH)
    unsigned short* __restrict__ q_out,        // (B*HEADS, N, DH) bf16
    unsigned short* __restrict__ k_out)        // (B*HEADS, N, DH) bf16
{
    const int lane = threadIdx.x & 31;
    const int wid  = threadIdx.x >> 5;
    const int half = lane >> 4;
    const int l16  = lane & 15;
    const int task = blockIdx.x * 8 + wid;     // 400 mblocks * 16 oblocks
    const int mblk = task >> 4;
    const int oblk = task & 15;
    const int m0   = mblk * 16;
    const int o0   = oblk * 64;

    v8f acc[4];
    const v8f zero8 = {0.f,0.f,0.f,0.f,0.f,0.f,0.f,0.f};
    #pragma unroll
    for (int t = 0; t < 4; ++t) acc[t] = zero8;

    for (int k0 = 0; k0 < C_IN; k0 += 32) {
        // A fragment: x rows m0+l16 (same row for both lane halves), 32 K vals
        Frag16 a;
        const unsigned short* arow = xb + (size_t)(m0 + l16) * C_IN + k0;
        #pragma unroll
        for (int j = 0; j < 8; ++j) {
            const int kl = ((j < 4) ? 0 : 16) + 8 * half + (j & 3) * 2;
            a.w[j] = *(const uint32_t*)(arow + kl);
        }
        // B fragments: B[c][o] = w[o][c]; lane=N(col o), K=(j<4?0:16)+8*half+2*(j&3)+s
        #pragma unroll
        for (int t = 0; t < 4; ++t) {
            Frag16 bf;
            const unsigned short* wrow = wb + (size_t)(o0 + t * 16 + l16) * C_IN + k0;
            #pragma unroll
            for (int j = 0; j < 8; ++j)
                bf.w[j] = *(const uint32_t*)(wrow + ((j < 4) ? 0 : 16) + 8 * half + (j & 3) * 2);
            acc[t] = wmma_bf16(a.v, bf.v, acc[t]);
        }
    }

    // Epilogue: bias + RoPE (+ scale for q), store bf16.
    const int  b    = m0 / NPOS;               // 16-row block never straddles b
    const bool is_q = (oblk < 8);
    const int  head = is_q ? oblk : (oblk - 8);
    unsigned short* outp = is_q ? q_out : k_out;
    const float scale = 0.125f;                // dh^-0.5, folded into q

    #pragma unroll
    for (int t = 0; t < 4; ++t) {
        const int d  = t * 16 + l16;
        const float bv = bias[o0 + t * 16 + l16];
        #pragma unroll
        for (int r = 0; r < 8; ++r) {
            const int m = m0 + r + 8 * half;
            const int n = m % NPOS;
            float val = acc[t][r] + bv;
            float partner = __shfl_xor(val, 1, 32);      // t[d^1] (+ its bias)
            // rotate_every_two: rot[even d] = -t[d+1], rot[odd d] = t[d-1]
            float rot = (d & 1) ? partner : -partner;
            float res = val * cosp[n * DH + d] + rot * sinp[n * DH + d];
            if (is_q) res *= scale;
            outp[(size_t)((b * HEADS + head) * NPOS + n) * DH + d] = f32_to_bf16(res);
        }
    }
}

// ---------------------------------------------------------------------------
// Kernel 2: flash attention per (b,head): S = q k^T (scale pre-folded),
// online softmax, O = P k  (reference uses k, not v).
// One wave owns 16 q rows; streams 32 k rows per step through LDS, staged
// via GLOBAL_LOAD_ASYNC_TO_LDS (ASYNCcnt) when the toolchain exposes it.
// 3200 wave tasks = 400 blocks x 8 waves. Output: (B, N, C) fp32.
// ---------------------------------------------------------------------------
__global__ __launch_bounds__(256) void flash_attn_kernel(
    const unsigned short* __restrict__ qb,   // (B*HEADS, N, DH) bf16
    const unsigned short* __restrict__ kb,   // (B*HEADS, N, DH) bf16
    float* __restrict__ out)                 // (B, N, C)
{
    __shared__ __align__(16) unsigned short kt[8][32 * DH];  // 8 waves * 4KB
    __shared__ __align__(16) unsigned short pt[8][16 * 32];  // 8 waves * 1KB

    const int lane = threadIdx.x & 31;
    const int wid  = threadIdx.x >> 5;
    const int half = lane >> 4;
    const int l16  = lane & 15;
    const int task = blockIdx.x * 8 + wid;   // 32 bh * 100 qblocks
    const int bh   = task / 100;
    const int qblk = task % 100;
    const int b    = bh >> 3;
    const int h    = bh & 7;

    const unsigned short* qbase = qb + (size_t)bh * NPOS * DH;
    const unsigned short* kbase = kb + (size_t)bh * NPOS * DH;

    // q A-fragments for d=[0,32) and d=[32,64)
    Frag16 qa[2];
    {
        const unsigned short* qrow = qbase + (size_t)(qblk * 16 + l16) * DH;
        #pragma unroll
        for (int f = 0; f < 2; ++f)
            #pragma unroll
            for (int j = 0; j < 8; ++j) {
                const int kl = ((j < 4) ? 0 : 16) + 8 * half + (j & 3) * 2;
                qa[f].w[j] = *(const uint32_t*)(qrow + f * 32 + kl);
            }
    }

    const v8f zero8 = {0.f,0.f,0.f,0.f,0.f,0.f,0.f,0.f};
    v8f acc[4];
    #pragma unroll
    for (int t = 0; t < 4; ++t) acc[t] = zero8;
    float mi[8], li[8];
    #pragma unroll
    for (int r = 0; r < 8; ++r) { mi[r] = -1e30f; li[r] = 0.f; }

    unsigned short* myk = kt[wid];
    unsigned short* myp = pt[wid];

    for (int mt = 0; mt < NPOS; mt += 32) {
        // --- Stage 32x64 bf16 k tile into LDS: lane owns one 128B row. ---
#if USE_ASYNC_LDS
        {
            // Prior DS reads of this tile must have landed before overwrite.
            asm volatile("s_wait_dscnt 0x0" ::: "memory");
            const AS1 char* g = (const AS1 char*)(kbase + (size_t)(mt + lane) * DH);
            AS3 char* l = (AS3 char*)(myk + lane * DH);
            #pragma unroll
            for (int i = 0; i < 8; ++i)
                __builtin_amdgcn_global_load_async_to_lds_b128(
                    (AS1 v4i*)(g + i * 16), (AS3 v4i*)(l + i * 16),
                    /*imm offset=*/0, /*cpol=*/0);
        }
#else
        {
            const uint4* src = (const uint4*)(kbase + (size_t)(mt + lane) * DH);
            uint4* dst = (uint4*)(myk + lane * DH);
            #pragma unroll
            for (int i = 0; i < 8; ++i) dst[i] = src[i];
        }
#endif
        // Prefetch next tile (global_prefetch_b8); uniform clamp -> no divergence.
        {
            const int nxt = (mt + 32 < NPOS) ? (mt + 32) : mt;
            __builtin_prefetch(kbase + (size_t)(nxt + lane) * DH, 0, 1);
        }
#if USE_ASYNC_LDS
        // Tile must be resident in LDS before the fragment gathers below.
#if __has_builtin(__builtin_amdgcn_s_wait_asynccnt)
        __builtin_amdgcn_s_wait_asynccnt(0);
#else
        asm volatile("s_wait_asynccnt 0x0" ::: "memory");
#endif
#endif

        // S tiles: S[n][m] = sum_d q[n][d] k[m][d]; B[K=d][N=mlocal] from LDS.
        v8f s[2];
        #pragma unroll
        for (int st = 0; st < 2; ++st) {
            Frag16 kf0, kf1;
            const unsigned short* krow = myk + (st * 16 + l16) * DH;
            #pragma unroll
            for (int j = 0; j < 8; ++j) {
                const int d0 = ((j < 4) ? 0 : 16) + 8 * half + (j & 3) * 2;
                kf0.w[j] = *(const uint32_t*)(krow + d0);
                kf1.w[j] = *(const uint32_t*)(krow + 32 + d0);
            }
            v8f c = zero8;
            c = wmma_bf16(qa[0].v, kf0.v, c);
            c = wmma_bf16(qa[1].v, kf1.v, c);
            s[st] = c;
        }

        // Online softmax; row M=r+8*half lives across the lane's 16-lane half.
        #pragma unroll
        for (int r = 0; r < 8; ++r) {
            float rm = fmaxf(s[0][r], s[1][r]);
            #pragma unroll
            for (int msk = 1; msk <= 8; msk <<= 1)
                rm = fmaxf(rm, __shfl_xor(rm, msk, 32));
            const float mnew  = fmaxf(mi[r], rm);
            const float alpha = __expf(mi[r] - mnew);
            const float p0 = __expf(s[0][r] - mnew);
            const float p1 = __expf(s[1][r] - mnew);
            float rs = p0 + p1;
            #pragma unroll
            for (int msk = 1; msk <= 8; msk <<= 1)
                rs += __shfl_xor(rs, msk, 32);
            li[r] = li[r] * alpha + rs;
            mi[r] = mnew;
            #pragma unroll
            for (int t = 0; t < 4; ++t) acc[t][r] *= alpha;
            // C-layout -> row-major bf16 P tile in LDS (for A-fragment reload)
            const int row = r + 8 * half;
            myp[row * 32 + l16]      = f32_to_bf16(p0);
            myp[row * 32 + 16 + l16] = f32_to_bf16(p1);
        }

        // P as A-fragment (16x32 over mlocal)
        Frag16 pa;
        {
            const unsigned short* prow = myp + l16 * 32;
            #pragma unroll
            for (int j = 0; j < 8; ++j) {
                const int kl = ((j < 4) ? 0 : 16) + 8 * half + (j & 3) * 2;
                pa.w[j] = *(const uint32_t*)(prow + kl);
            }
        }

        // O += P x K: B[K=mlocal][N=d] from the same LDS k tile.
        #pragma unroll
        for (int dt = 0; dt < 4; ++dt) {
            Frag16 kf;
            const int d = dt * 16 + l16;
            #pragma unroll
            for (int j = 0; j < 8; ++j) {
                const int m0l = ((j < 4) ? 0 : 16) + 8 * half + (j & 3) * 2;
                const uint32_t e0 = myk[m0l * DH + d];
                const uint32_t e1 = myk[(m0l + 1) * DH + d];
                kf.w[j] = e0 | (e1 << 16);
            }
            acc[dt] = wmma_bf16(pa.v, kf.v, acc[dt]);
        }
    }

    // Normalize and scatter to (B, H, W, C) = (b, n, h*64+d)
    #pragma unroll
    for (int dt = 0; dt < 4; ++dt) {
        #pragma unroll
        for (int r = 0; r < 8; ++r) {
            const int n = qblk * 16 + r + 8 * half;
            const int d = dt * 16 + l16;
            out[(size_t)(b * NPOS + n) * C_IN + h * DH + d] = acc[dt][r] / li[r];
        }
    }
}

// ---------------------------------------------------------------------------
// Launch
// ---------------------------------------------------------------------------
extern "C" void kernel_launch(void* const* d_in, const int* in_sizes, int n_in,
                              void* d_out, int out_size, void* d_ws, size_t ws_size,
                              hipStream_t stream) {
    (void)in_sizes; (void)n_in; (void)out_size; (void)ws_size;
    const float* x     = (const float*)d_in[0];  // (4,40,40,512)
    const float* sinp  = (const float*)d_in[1];  // (40,40,64)
    const float* cosp  = (const float*)d_in[2];  // (40,40,64)
    const float* w_qkv = (const float*)d_in[3];  // (1536,512)
    const float* b_qkv = (const float*)d_in[4];  // (1536,)
    float* out = (float*)d_out;

    const size_t n_x  = (size_t)BATCH * NPOS * C_IN;   // 3,276,800
    const size_t n_w  = (size_t)QK_O * C_IN;           //   524,288
    const size_t n_qk = (size_t)BATCH * HEADS * NPOS * DH;

    char* ws = (char*)d_ws;
    unsigned short* x_bf = (unsigned short*)ws;                 ws += n_x  * 2;
    unsigned short* w_bf = (unsigned short*)ws;                 ws += n_w  * 2;
    unsigned short* q_bf = (unsigned short*)ws;                 ws += n_qk * 2;
    unsigned short* k_bf = (unsigned short*)ws;

    cvt_bf16_kernel<<<2048, 256, 0, stream>>>(x, x_bf, (int)n_x);
    cvt_bf16_kernel<<<1024, 256, 0, stream>>>(w_qkv, w_bf, (int)n_w);
    // 400 mblocks * 16 oblocks = 6400 wave tasks, 8 waves/block
    qk_proj_rope_kernel<<<800, 256, 0, stream>>>(x_bf, w_bf, b_qkv, sinp, cosp,
                                                 q_bf, k_bf);
    // 32 (b,h) * 100 qblocks = 3200 wave tasks, 8 waves/block
    flash_attn_kernel<<<400, 256, 0, stream>>>(q_bf, k_bf, out);
}
